// MultiHeadAttention_39986145526247
// MI455X (gfx1250) — compile-verified, matrix operand order checked by reading, not run on hardware
//
#include <hip/hip_runtime.h>

// Problem constants (from reference)
#define B_  4
#define T_  2048
#define C_  2048
#define H_  16
#define HD_ 128
#define C3_ (3 * C_)

typedef __bf16 bf16_t;
typedef bf16_t v16bf __attribute__((ext_vector_type(16)));
typedef float  v8f   __attribute__((ext_vector_type(8)));

union F8   { v8f v; float f[8]; };
union FRAG { v16bf v; uint4 q[2]; };

// float -> bf16, round-to-nearest-even (pure integer ops; no __bf16 cast needed)
static __device__ __forceinline__ bf16_t f2bf(float f) {
  unsigned u = __float_as_uint(f);
  u += 0x7FFFu + ((u >> 16) & 1u);
  unsigned short s = (unsigned short)(u >> 16);
  union { unsigned short u; bf16_t b; } cv; cv.u = s;
  return cv.b;
}

// Load one 16x32 bf16 A/B fragment for v_wmma_f32_16x16x32_bf16.
// Per ISA 7.12.2: lane holds row (m = lane%16), K values packed 2/VGPR:
//   VGPR v<4 : K = g*8 + 2v      (g = lane/16)
//   VGPR v>=4: K = 16 + g*8 + 2(v-4)
// => two contiguous 16B loads from the 64B row at byte offsets g*16 and 32+g*16.
static __device__ __forceinline__ v16bf load_frag(const bf16_t* row, int g) {
  FRAG r;
  const char* p = (const char*)row + g * 16;
  r.q[0] = *(const uint4*)(p);
  r.q[1] = *(const uint4*)(p + 32);
  return r.v;
}

// CDNA5 direct global->LDS async copy (ASYNCcnt path), 16B per lane.
// lds  : wave-relative LDS byte offset (low 32 bits of generic shared pointer)
// gptr : 64-bit global address
static __device__ __forceinline__ void async_copy_b128(unsigned lds, const void* gptr) {
  asm volatile("global_load_async_to_lds_b128 %0, %1, off"
               :: "v"(lds), "v"(gptr)
               : "memory");
}
static __device__ __forceinline__ void async_wait0() {
  asm volatile("s_wait_asynccnt 0x0" ::: "memory");
}

// ---------------------------------------------------------------------------
// x (f32) -> bf16, elementwise (4 per thread)
// ---------------------------------------------------------------------------
__global__ __launch_bounds__(256) void cvt_f32_bf16(const float* __restrict__ in,
                                                    bf16_t* __restrict__ out) {
  size_t i = ((size_t)blockIdx.x * blockDim.x + threadIdx.x) * 4;
  float4 v = *(const float4*)(in + i);
  out[i + 0] = f2bf(v.x);
  out[i + 1] = f2bf(v.y);
  out[i + 2] = f2bf(v.z);
  out[i + 3] = f2bf(v.w);
}

// ---------------------------------------------------------------------------
// Wt[n][k] = bf16(W[k][n])  (32x32 LDS tile transpose)
// ---------------------------------------------------------------------------
__global__ __launch_bounds__(256) void transpose_cvt(const float* __restrict__ W,
                                                     bf16_t* __restrict__ Wt,
                                                     int K, int N) {
  __shared__ float tile[32][33];
  const int n0 = blockIdx.x * 32, k0 = blockIdx.y * 32;
  const int tx = threadIdx.x, ty = threadIdx.y;
#pragma unroll
  for (int i = 0; i < 4; ++i)
    tile[ty + i * 8][tx] = W[(size_t)(k0 + ty + i * 8) * N + n0 + tx];
  __syncthreads();
#pragma unroll
  for (int i = 0; i < 4; ++i)
    Wt[(size_t)(n0 + ty + i * 8) * K + k0 + tx] = f2bf(tile[tx][ty + i * 8]);
}

// ---------------------------------------------------------------------------
// Vt[b][h][d][t] = qkv[b][t][2C + h*hd + d]   (bf16 32x32 tile transpose)
// ---------------------------------------------------------------------------
__global__ __launch_bounds__(256) void vtrans(const bf16_t* __restrict__ qkv,
                                              bf16_t* __restrict__ vt) {
  __shared__ bf16_t tile[32][34];
  const int bh = blockIdx.z, b = bh >> 4, h = bh & 15;
  const int t0 = blockIdx.x * 32, d0 = blockIdx.y * 32;
  const int tx = threadIdx.x, ty = threadIdx.y;
  const bf16_t* src = qkv + (size_t)b * T_ * C3_ + 2 * C_ + h * HD_ + d0;
#pragma unroll
  for (int i = 0; i < 4; ++i)
    tile[ty + i * 8][tx] = src[(size_t)(t0 + ty + i * 8) * C3_ + tx];
  __syncthreads();
  bf16_t* dst = vt + (size_t)bh * HD_ * T_ + (size_t)d0 * T_ + t0;
#pragma unroll
  for (int i = 0; i < 4; ++i)
    dst[(size_t)(ty + i * 8) * T_ + tx] = tile[tx][ty + i * 8];
}

// ---------------------------------------------------------------------------
// GEMM: out[M,N] = A[M,K](bf16) * Bt[N,K](bf16)^T + bias
// Workgroup: 256 threads = 8 waves (2x4 grid), tile 64x256, 32x64 per wave.
// Tiles staged into padded LDS via GLOBAL_LOAD_ASYNC_TO_LDS_B128 (ASYNCcnt),
// then 8 WMMAs per wave per 32-wide K chunk.
// ---------------------------------------------------------------------------
__global__ __launch_bounds__(256) void gemm_wmma(const bf16_t* __restrict__ A,
                                                 const bf16_t* __restrict__ Bt,
                                                 const float* __restrict__ bias,
                                                 void* __restrict__ out,
                                                 int M, int N, int K, int out_bf16) {
  __shared__ bf16_t sA[64 * 40];    // 40 = 32 + 8 bf16 pad (keeps 16B alignment)
  __shared__ bf16_t sB[256 * 40];
  const int tid  = threadIdx.x;
  const int lane = tid & 31, wave = tid >> 5;
  const int wm = wave >> 2, wn = wave & 3;           // 2x4 wave grid
  const int g = lane >> 4, ln = lane & 15;
  const int bm = blockIdx.y * 64, bn = blockIdx.x * 256;

  F8 acc[2][4];
#pragma unroll
  for (int mi = 0; mi < 2; ++mi)
#pragma unroll
    for (int ni = 0; ni < 4; ++ni)
#pragma unroll
      for (int r = 0; r < 8; ++r) acc[mi][ni].f[r] = 0.f;

  // Per-thread staging assignments (16B units)
  const int arow = tid >> 2, aseg = tid & 3;

  for (int k0 = 0; k0 < K; k0 += 32) {
    // A tile: 64 rows x 32 bf16 ; one async b128 per thread
    async_copy_b128((unsigned)(size_t)(sA + arow * 40 + aseg * 8),
                    A + (size_t)(bm + arow) * K + k0 + aseg * 8);
    // B tile: 256 rows x 32 bf16 ; four async b128 per thread
#pragma unroll
    for (int i = 0; i < 4; ++i) {
      int idx = tid + i * 256, row = idx >> 2, seg = idx & 3;
      async_copy_b128((unsigned)(size_t)(sB + row * 40 + seg * 8),
                      Bt + (size_t)(bn + row) * K + k0 + seg * 8);
    }
    async_wait0();
    __syncthreads();

    v16bf af[2], bfr[4];
#pragma unroll
    for (int mi = 0; mi < 2; ++mi)
      af[mi] = load_frag(sA + (wm * 32 + mi * 16 + ln) * 40, g);
#pragma unroll
    for (int ni = 0; ni < 4; ++ni)
      bfr[ni] = load_frag(sB + (wn * 64 + ni * 16 + ln) * 40, g);
#pragma unroll
    for (int mi = 0; mi < 2; ++mi)
#pragma unroll
      for (int ni = 0; ni < 4; ++ni)
        acc[mi][ni].v = __builtin_amdgcn_wmma_f32_16x16x32_bf16(
            false, af[mi], false, bfr[ni], (short)0, acc[mi][ni].v, false, false);
    __syncthreads();
  }

  // Epilogue: D layout => lane holds n = lane%16, rows m = r + 8*(lane/16)
#pragma unroll
  for (int mi = 0; mi < 2; ++mi)
#pragma unroll
    for (int ni = 0; ni < 4; ++ni) {
      const int row0 = bm + wm * 32 + mi * 16 + g * 8;
      const int col  = bn + wn * 64 + ni * 16 + ln;
      const float bv = bias[col];
      if (out_bf16) {
        bf16_t* o = (bf16_t*)out;
#pragma unroll
        for (int r = 0; r < 8; ++r)
          o[(size_t)(row0 + r) * N + col] = f2bf(acc[mi][ni].f[r] + bv);
      } else {
        float* o = (float*)out;
#pragma unroll
        for (int r = 0; r < 8; ++r)
          o[(size_t)(row0 + r) * N + col] = acc[mi][ni].f[r] + bv;
      }
    }
}

// ---------------------------------------------------------------------------
// Flash attention: one wave per 16-query block per (b,h).
// Q kept in registers (4 A-frags). Per 32-key step:
//   S = Q*K^T (8 WMMA), online softmax (shfl_xor row reductions),
//   P staged via LDS (D->A layout transpose), O += P*V (8 WMMA).
// ---------------------------------------------------------------------------
__global__ __launch_bounds__(32) void attn_wmma(const bf16_t* __restrict__ qkv,
                                                const bf16_t* __restrict__ vt,
                                                bf16_t* __restrict__ y) {
  __shared__ bf16_t sP[16 * 40];
  const int lane = threadIdx.x;
  const int g = lane >> 4, ln = lane & 15;
  const int bh = blockIdx.y, b = bh >> 4, h = bh & 15;
  const int q0 = blockIdx.x * 16;

  const bf16_t* qbase = qkv + ((size_t)b * T_ + q0) * C3_ + h * HD_;
  const bf16_t* kbase = qkv + (size_t)b * T_ * C3_ + C_ + h * HD_;
  const bf16_t* vbase = vt + (size_t)bh * HD_ * T_;

  v16bf qf[4];
#pragma unroll
  for (int c = 0; c < 4; ++c)
    qf[c] = load_frag(qbase + (size_t)ln * C3_ + c * 32, g);

  F8 o[8];
  float mrow[8], lrow[8];
#pragma unroll
  for (int db = 0; db < 8; ++db)
#pragma unroll
    for (int r = 0; r < 8; ++r) o[db].f[r] = 0.f;
#pragma unroll
  for (int r = 0; r < 8; ++r) { mrow[r] = -1e30f; lrow[r] = 0.f; }

  const float scale = 0.08838834764831845f;  // 1/sqrt(128)

  for (int kb = 0; kb < T_; kb += 32) {
    F8 s[2];
#pragma unroll
    for (int nb = 0; nb < 2; ++nb) {
#pragma unroll
      for (int r = 0; r < 8; ++r) s[nb].f[r] = 0.f;
#pragma unroll
      for (int c = 0; c < 4; ++c) {
        v16bf kf = load_frag(kbase + (size_t)(kb + nb * 16 + ln) * C3_ + c * 32, g);
        s[nb].v = __builtin_amdgcn_wmma_f32_16x16x32_bf16(
            false, qf[c], false, kf, (short)0, s[nb].v, false, false);
      }
    }
    // Online softmax update per row m = r + 8g (shared by 16 lanes of a group)
#pragma unroll
    for (int r = 0; r < 8; ++r) {
      float v0 = s[0].f[r] * scale, v1 = s[1].f[r] * scale;
      float vmax = fmaxf(v0, v1);
#pragma unroll
      for (int off = 8; off >= 1; off >>= 1)
        vmax = fmaxf(vmax, __shfl_xor(vmax, off, 32));
      const float mnew  = fmaxf(mrow[r], vmax);
      const float alpha = __expf(mrow[r] - mnew);
      mrow[r] = mnew;
      const float p0 = __expf(v0 - mnew), p1 = __expf(v1 - mnew);
      sP[(r + 8 * g) * 40 + ln]      = f2bf(p0);
      sP[(r + 8 * g) * 40 + 16 + ln] = f2bf(p1);
      float ps = p0 + p1;
#pragma unroll
      for (int off = 8; off >= 1; off >>= 1)
        ps += __shfl_xor(ps, off, 32);
      lrow[r] = lrow[r] * alpha + ps;
#pragma unroll
      for (int db = 0; db < 8; ++db) o[db].f[r] *= alpha;
    }
    __syncthreads();
    v16bf pf = load_frag(sP + ln * 40, g);  // P as 16x32 A fragment
#pragma unroll
    for (int db = 0; db < 8; ++db) {
      v16bf vf = load_frag(vbase + (size_t)(db * 16 + ln) * T_ + kb, g);
      o[db].v = __builtin_amdgcn_wmma_f32_16x16x32_bf16(
          false, pf, false, vf, (short)0, o[db].v, false, false);
    }
    __syncthreads();
  }

  bf16_t* ybase = y + ((size_t)b * T_ + q0) * C_ + h * HD_;
#pragma unroll
  for (int db = 0; db < 8; ++db)
#pragma unroll
    for (int r = 0; r < 8; ++r)
      ybase[(size_t)(r + 8 * g) * C_ + db * 16 + ln] = f2bf(o[db].f[r] / lrow[r]);
}

// ---------------------------------------------------------------------------
// Host launch
// ---------------------------------------------------------------------------
extern "C" void kernel_launch(void* const* d_in, const int* in_sizes, int n_in,
                              void* d_out, int out_size, void* d_ws, size_t ws_size,
                              hipStream_t stream) {
  const float* x      = (const float*)d_in[0];
  const float* w_attn = (const float*)d_in[1];
  const float* b_attn = (const float*)d_in[2];
  const float* w_proj = (const float*)d_in[3];
  const float* b_proj = (const float*)d_in[4];

  char* ws = (char*)d_ws;
  bf16_t* xb  = (bf16_t*)(ws);                    // 33,554,432 B : x bf16
  bf16_t* wat = (bf16_t*)(ws + 33554432);         // 25,165,824 B : w_attn^T bf16
  bf16_t* wpt = (bf16_t*)(ws + 58720256);         //  8,388,608 B : w_proj^T bf16
  bf16_t* qkv = (bf16_t*)(ws + 67108864);         // 100,663,296 B: qkv bf16
  bf16_t* vt  = (bf16_t*)(ws + 167772160);        // 33,554,432 B : V^T per (b,h)
  bf16_t* y   = (bf16_t*)(ws + 201326592);        // 33,554,432 B : attn out bf16

  // 1) conversions / transposes
  cvt_f32_bf16<<<16384, 256, 0, stream>>>(x, xb);  // 16384*256*4 = B*T*C
  transpose_cvt<<<dim3(C3_ / 32, C_ / 32), dim3(32, 8), 0, stream>>>(w_attn, wat, C_, C3_);
  transpose_cvt<<<dim3(C_ / 32, C_ / 32), dim3(32, 8), 0, stream>>>(w_proj, wpt, C_, C_);

  // 2) QKV projection: [8192,2048] x [2048,6144] -> bf16 qkv (+ bias)
  gemm_wmma<<<dim3(C3_ / 256, (B_ * T_) / 64), 256, 0, stream>>>(
      xb, wat, b_attn, qkv, B_ * T_, C3_, C_, 1);

  // 3) V transpose for PV fragment loads
  vtrans<<<dim3(T_ / 32, HD_ / 32, B_ * H_), dim3(32, 8), 0, stream>>>(qkv, vt);

  // 4) flash attention
  attn_wmma<<<dim3(T_ / 16, B_ * H_), 32, 0, stream>>>(qkv, vt, y);

  // 5) output projection: [8192,2048] x [2048,2048] -> f32 d_out (+ bias)
  gemm_wmma<<<dim3(C_ / 256, (B_ * T_) / 64), 256, 0, stream>>>(
      y, wpt, b_proj, d_out, B_ * T_, C_, C_, 0);
}
